// SEALAttention_24730421690607
// MI455X (gfx1250) — matrix operand order repaired and checked by reading, not compiled, not run on hardware
//
#include <hip/hip_runtime.h>
#include <cstdint>

#define DMODEL 1024
#define SEGLEN 256
#define ATT_SCALE 0.125f   // 1/sqrt(64) per reference (HEAD_DIM=64)

typedef __attribute__((ext_vector_type(16))) _Float16 v16h;
typedef __attribute__((ext_vector_type(8)))  _Float16 v8h;
typedef __attribute__((ext_vector_type(8)))  float    v8f;
typedef __attribute__((ext_vector_type(4)))  int      v4i;

__device__ __forceinline__ v8f wmma32(v16h a, v16h b, v8f c) {
  return __builtin_amdgcn_wmma_f32_16x16x32_f16(false, a, false, b, (short)0, c, false, false);
}

// A fragment 16x32 (f16): lane m = lane%16, h = lane/16.
// VGPRs 0-3: K = h*8 .. h*8+7 ; VGPRs 4-7: K = 16+h*8 .. 16+h*8+7
__device__ __forceinline__ v16h load_a(const _Float16* row0base, int ld, int k0, int lane) {
  const _Float16* p = row0base + (size_t)(lane & 15) * ld + k0 + ((lane >> 4) << 3);
  union { v16h v; v8h h[2]; } u;
  u.h[0] = *(const v8h*)p;
  u.h[1] = *(const v8h*)(p + 16);
  return u.v;
}

// B fragment 32x16 (f16) from row-major source where row r = output column n:
// lane n = lane%16, h = lane/16; VGPRs 0-3: K = h*8 .. h*8+7 ; VGPRs 4-7: K = 16+h*8 .. 16+h*8+7
__device__ __forceinline__ v16h load_b(const _Float16* base, int ld, int col0, int k0, int lane) {
  const _Float16* p = base + (size_t)(col0 + (lane & 15)) * ld + k0 + ((lane >> 4) << 3);
  union { v16h v; v8h h[2]; } u;
  u.h[0] = *(const v8h*)p;
  u.h[1] = *(const v8h*)(p + 16);
  return u.v;
}

// ---------------------------------------------------------------- async-to-LDS probe
// Not launched. Compile-time probe for the gfx1250 async copy path (ASYNCcnt).
// Round-2 diagnostic showed param 0 is `v4i*` (width encoded in pointee type).
#if defined(__HIP_DEVICE_COMPILE__) && defined(__has_builtin)
#if __has_builtin(__builtin_amdgcn_global_load_async_to_lds_b128)
#define HAVE_ASYNC_B128 1
#endif
#endif

#ifdef HAVE_ASYNC_B128
__global__ void async_probe_kernel(const _Float16* __restrict__ g, float* __restrict__ out) {
  __shared__ _Float16 buf[2048];
  v4i* lptr = (v4i*)(uintptr_t)(&buf[(threadIdx.x & 255) * 8]);
  v4i* gptr = (v4i*)(uintptr_t)(g + (threadIdx.x & 255) * 8);
  __builtin_amdgcn_global_load_async_to_lds_b128(gptr, lptr, 0, 0);
#if __has_builtin(__builtin_amdgcn_s_wait_asynccnt)
  __builtin_amdgcn_s_wait_asynccnt(0);
#else
  asm volatile("s_wait_asynccnt 0" ::: "memory");
#endif
  __syncthreads();
  out[threadIdx.x] = (float)buf[threadIdx.x & 2047];
}
#endif

// ---------------------------------------------------------------- convert
__global__ void convert_kernel(const float* __restrict__ x,
                               const float* __restrict__ wq, const float* __restrict__ wk,
                               const float* __restrict__ wv, const float* __restrict__ wo,
                               _Float16* __restrict__ xh,
                               _Float16* __restrict__ wqh, _Float16* __restrict__ wkh,
                               _Float16* __restrict__ wvh, _Float16* __restrict__ woh) {
  const size_t NX = (size_t)4 * 4096 * DMODEL;      // 16777216
  const size_t NW = (size_t)DMODEL * DMODEL;        // 1048576
  size_t stride = (size_t)gridDim.x * blockDim.x;
  size_t i0 = (size_t)blockIdx.x * blockDim.x + threadIdx.x;
  for (size_t i = i0; i < NX; i += stride) xh[i] = (_Float16)x[i];
  for (size_t i = i0; i < NW; i += stride) {
    wqh[i] = (_Float16)wq[i];
    wkh[i] = (_Float16)wk[i];
    wvh[i] = (_Float16)wv[i];
    woh[i] = (_Float16)wo[i];
  }
}

// ---------------------------------------------------------------- QKV GEMM
// C[m,n] = sum_d X[m,d] * W[n,d] + b[n]   (einsum 'md,nd->mn')
// Per wave: 32 x 32 output tile for Q, K, V simultaneously.
// Per k-step: 2 A frags + 6 B frags -> 12 WMMAs (each B frag reused across both A frags).
// Q, K stored row-major f16. V stored transposed per segment: vt[seg][e][s].
__global__ void __launch_bounds__(256) qkv_gemm(const _Float16* __restrict__ xh,
                         const _Float16* __restrict__ wqh, const _Float16* __restrict__ wkh,
                         const _Float16* __restrict__ wvh,
                         const float* __restrict__ bq, const float* __restrict__ bk,
                         const float* __restrict__ bv,
                         _Float16* __restrict__ qh, _Float16* __restrict__ kh,
                         _Float16* __restrict__ vt) {
  const int K = DMODEL, N = DMODEL;
  int wave = (int)((blockIdx.x * blockDim.x + threadIdx.x) >> 5);
  int lane = threadIdx.x & 31;
  const int NT2 = N / 32;                       // 32
  int mt  = wave / NT2;                         // 0..511
  int nt2 = wave % NT2;                         // 0..31
  int row0 = mt * 32, col0 = nt2 * 32;

  v8f cq[2][2], ck[2][2], cv[2][2];
#pragma unroll
  for (int r = 0; r < 2; r++)
#pragma unroll
    for (int i = 0; i < 2; i++) { cq[r][i] = v8f{}; ck[r][i] = v8f{}; cv[r][i] = v8f{}; }

  const _Float16* arow0 = xh + (size_t)row0 * K;
  const _Float16* arow1 = arow0 + (size_t)16 * K;
  for (int k0 = 0; k0 < K; k0 += 32) {
    v16h a0  = load_a(arow0, K, k0, lane);
    v16h a1  = load_a(arow1, K, k0, lane);
    v16h bq0 = load_b(wqh, K, col0,      k0, lane);
    v16h bq1 = load_b(wqh, K, col0 + 16, k0, lane);
    v16h bk0 = load_b(wkh, K, col0,      k0, lane);
    v16h bk1 = load_b(wkh, K, col0 + 16, k0, lane);
    v16h bv0 = load_b(wvh, K, col0,      k0, lane);
    v16h bv1 = load_b(wvh, K, col0 + 16, k0, lane);
    cq[0][0] = wmma32(a0, bq0, cq[0][0]);  cq[0][1] = wmma32(a0, bq1, cq[0][1]);
    cq[1][0] = wmma32(a1, bq0, cq[1][0]);  cq[1][1] = wmma32(a1, bq1, cq[1][1]);
    ck[0][0] = wmma32(a0, bk0, ck[0][0]);  ck[0][1] = wmma32(a0, bk1, ck[0][1]);
    ck[1][0] = wmma32(a1, bk0, ck[1][0]);  ck[1][1] = wmma32(a1, bk1, ck[1][1]);
    cv[0][0] = wmma32(a0, bv0, cv[0][0]);  cv[0][1] = wmma32(a0, bv1, cv[0][1]);
    cv[1][0] = wmma32(a1, bv0, cv[1][0]);  cv[1][1] = wmma32(a1, bv1, cv[1][1]);
  }

  int n0 = col0 + (lane & 15);
  int n1 = n0 + 16;
  float bQ0 = bq[n0], bQ1 = bq[n1];
  float bK0 = bk[n0], bK1 = bk[n1];
  float bV0 = bv[n0], bV1 = bv[n1];
  int seg = row0 >> 8;                          // all 32 rows in same segment
  _Float16* vtseg = vt + (size_t)seg * DMODEL * SEGLEN;
#pragma unroll
  for (int rh = 0; rh < 2; rh++) {
    int rbase = row0 + rh * 16 + ((lane >> 4) << 3);
#pragma unroll
    for (int j = 0; j < 8; j++) {
      int r = rbase + j;
      qh[(size_t)r * N + n0] = (_Float16)(cq[rh][0][j] + bQ0);
      qh[(size_t)r * N + n1] = (_Float16)(cq[rh][1][j] + bQ1);
      kh[(size_t)r * N + n0] = (_Float16)(ck[rh][0][j] + bK0);
      kh[(size_t)r * N + n1] = (_Float16)(ck[rh][1][j] + bK1);
      int sl = r & (SEGLEN - 1);                // row within segment
      vtseg[(size_t)n0 * SEGLEN + sl] = (_Float16)(cv[rh][0][j] + bV0);
      vtseg[(size_t)n1 * SEGLEN + sl] = (_Float16)(cv[rh][1][j] + bV1);
    }
  }
}

// ---------------------------------------------------------------- global attention (tiny)
// gout[b*16+n][e] = sum_m softmax_m(SCALE * Qg[n].Kg[m]) * Vg[m][e]
__global__ void global_attn(const _Float16* __restrict__ qh, const _Float16* __restrict__ kh,
                            const _Float16* __restrict__ vt, float* __restrict__ gout) {
  int b = blockIdx.x;                 // 0..3
  int tid = threadIdx.x;              // 0..255
  __shared__ float attn[16][16];
  int n = tid >> 4, m = tid & 15;
  const _Float16* qrow = qh + (size_t)(b * 4096 + n * SEGLEN) * DMODEL;
  const _Float16* krow = kh + (size_t)(b * 4096 + m * SEGLEN) * DMODEL;
  float acc = 0.f;
  for (int e = 0; e < DMODEL; e++) acc += (float)qrow[e] * (float)krow[e];
  acc *= ATT_SCALE;
  float mx = acc;
#pragma unroll
  for (int s = 8; s >= 1; s >>= 1) mx = fmaxf(mx, __shfl_xor(mx, s, 32));
  float ex = __expf(acc - mx);
  float sm = ex;
#pragma unroll
  for (int s = 8; s >= 1; s >>= 1) sm += __shfl_xor(sm, s, 32);
  attn[n][m] = ex / sm;
  __syncthreads();
  for (int idx = tid; idx < 16 * DMODEL; idx += 256) {
    int nn = idx >> 10, e = idx & (DMODEL - 1);
    float o = 0.f;
#pragma unroll
    for (int mm = 0; mm < 16; mm++) {
      // Vg[mm][e] = vt[(b*16+mm)][e][s=0]
      o += attn[nn][mm] * (float)vt[(((size_t)(b * 16 + mm) << 10) + e) << 8];
    }
    gout[((size_t)(b * 16 + nn) << 10) + e] = o;
  }
}

// ---------------------------------------------------------------- local attention
// One block = 128 query rows of one segment (8 waves x 16 rows). grid = 128.
__global__ void __launch_bounds__(256) local_attn(const _Float16* __restrict__ qh,
                          const _Float16* __restrict__ kh, const _Float16* __restrict__ vt,
                          const float* __restrict__ gout, _Float16* __restrict__ oh) {
  int seg  = blockIdx.x >> 1;                  // 0..63
  int half = blockIdx.x & 1;
  int wave = threadIdx.x >> 5;                 // 0..7
  int lane = threadIdx.x & 31;
  int srow = half * 128 + wave * 16;           // row within segment
  size_t rowg = (size_t)seg * SEGLEN + srow;   // global token row

  __shared__ _Float16 attn_lds[8][16 * SEGLEN];   // 64 KB

  // ---- scores: S[t] holds 16x16 tile at columns t*16..t*16+15, f32
  v8f S[16];
#pragma unroll
  for (int t = 0; t < 16; t++) S[t] = v8f{};
  const _Float16* qrow = qh + rowg * DMODEL;
  const _Float16* kseg = kh + (size_t)seg * SEGLEN * DMODEL;
  for (int k0 = 0; k0 < DMODEL; k0 += 32) {
    v16h a = load_a(qrow, DMODEL, k0, lane);
#pragma unroll
    for (int t = 0; t < 16; t++) {
      v16h bf = load_b(kseg, DMODEL, t * 16, k0, lane);
      S[t] = wmma32(a, bf, S[t]);
    }
  }
#pragma unroll
  for (int t = 0; t < 16; t++) S[t] *= ATT_SCALE;

  // ---- softmax over 256 columns, rows live in 16-lane halves (C layout)
#pragma unroll
  for (int j = 0; j < 8; j++) {
    float mx = -3.0e38f;
#pragma unroll
    for (int t = 0; t < 16; t++) mx = fmaxf(mx, S[t][j]);
#pragma unroll
    for (int s = 8; s >= 1; s >>= 1) mx = fmaxf(mx, __shfl_xor(mx, s, 32));
    float sm = 0.f;
#pragma unroll
    for (int t = 0; t < 16; t++) { float e = __expf(S[t][j] - mx); S[t][j] = e; sm += e; }
#pragma unroll
    for (int s = 8; s >= 1; s >>= 1) sm += __shfl_xor(sm, s, 32);
    float inv = 1.0f / sm;
#pragma unroll
    for (int t = 0; t < 16; t++) S[t][j] *= inv;
  }

  // ---- D-layout -> A-layout via LDS (wave-private 8 KB region, no barrier needed)
  {
    _Float16* ab = &attn_lds[wave][0];
    int c0 = lane & 15, rh = (lane >> 4) << 3;
#pragma unroll
    for (int t = 0; t < 16; t++)
#pragma unroll
      for (int j = 0; j < 8; j++)
        ab[(size_t)(rh + j) * SEGLEN + t * 16 + c0] = (_Float16)S[t][j];
  }

  v16h A8[8];
  {
    const _Float16* ab = &attn_lds[wave][0];
#pragma unroll
    for (int tc = 0; tc < 8; tc++) {
      const _Float16* p = ab + (size_t)(lane & 15) * SEGLEN + tc * 32 + ((lane >> 4) << 3);
      union { v16h v; v8h h[2]; } u;
      u.h[0] = *(const v8h*)p;
      u.h[1] = *(const v8h*)(p + 16);
      A8[tc] = u.v;
    }
  }

  // ---- local = attn @ V  (B frags contiguous thanks to transposed V), + global_out
  const _Float16* vtseg = vt + (size_t)seg * DMODEL * SEGLEN;
  const float* grow = gout + ((size_t)seg << 10);
  for (int et = 0; et < 64; et++) {
    int col0 = et * 16;
    v8f c = {};
#pragma unroll
    for (int tc = 0; tc < 8; tc++) {
      v16h bf = load_b(vtseg, SEGLEN, col0, tc * 32, lane);
      c = wmma32(A8[tc], bf, c);
    }
    int col = col0 + (lane & 15);
    float g = grow[col];
    int rbase = srow + ((lane >> 4) << 3);
#pragma unroll
    for (int j = 0; j < 8; j++) {
      size_t r = (size_t)seg * SEGLEN + rbase + j;
      oh[r * DMODEL + col] = (_Float16)(c[j] + g);
    }
  }
}

// ---------------------------------------------------------------- output GEMM (f32 out)
// out[m,n] = sum_d oh[m,d] * Wo[n,d] + bo[n];  per wave: 32 x 64 tile
// Per k-step: 2 A frags + 4 B frags -> 8 WMMAs.
__global__ void __launch_bounds__(256) out_gemm(const _Float16* __restrict__ oh,
                        const _Float16* __restrict__ woh, const float* __restrict__ bo,
                        float* __restrict__ out) {
  const int K = DMODEL, N = DMODEL;
  int wave = (int)((blockIdx.x * blockDim.x + threadIdx.x) >> 5);
  int lane = threadIdx.x & 31;
  const int NG = N / 64;                        // 16
  int mt = wave / NG, ng = wave % NG;           // mt 0..511
  int row0 = mt * 32, col0 = ng * 64;
  v8f c[2][4];
#pragma unroll
  for (int r = 0; r < 2; r++)
#pragma unroll
    for (int i = 0; i < 4; i++) c[r][i] = v8f{};
  const _Float16* arow0 = oh + (size_t)row0 * K;
  const _Float16* arow1 = arow0 + (size_t)16 * K;
  for (int k0 = 0; k0 < K; k0 += 32) {
    v16h a0 = load_a(arow0, K, k0, lane);
    v16h a1 = load_a(arow1, K, k0, lane);
#pragma unroll
    for (int i = 0; i < 4; i++) {
      v16h bf = load_b(woh, K, col0 + i * 16, k0, lane);
      c[0][i] = wmma32(a0, bf, c[0][i]);
      c[1][i] = wmma32(a1, bf, c[1][i]);
    }
  }
#pragma unroll
  for (int rh = 0; rh < 2; rh++) {
    int rbase = row0 + rh * 16 + ((lane >> 4) << 3);
#pragma unroll
    for (int i = 0; i < 4; i++) {
      int col = col0 + i * 16 + (lane & 15);
      float bias = bo[col];
#pragma unroll
      for (int j = 0; j < 8; j++)
        out[(size_t)(rbase + j) * N + col] = c[rh][i][j] + bias;
    }
  }
}

// ---------------------------------------------------------------- launch
extern "C" void kernel_launch(void* const* d_in, const int* in_sizes, int n_in,
                              void* d_out, int out_size, void* d_ws, size_t ws_size,
                              hipStream_t stream) {
  const float* x  = (const float*)d_in[0];
  const float* Wq = (const float*)d_in[1];
  const float* bq = (const float*)d_in[2];
  const float* Wk = (const float*)d_in[3];
  const float* bk = (const float*)d_in[4];
  const float* Wv = (const float*)d_in[5];
  const float* bv = (const float*)d_in[6];
  const float* Wo = (const float*)d_in[7];
  const float* bo = (const float*)d_in[8];

  const size_t NX = (size_t)4 * 4096 * DMODEL;   // 16777216
  const size_t NW = (size_t)DMODEL * DMODEL;     // 1048576

  _Float16* xh  = (_Float16*)d_ws;
  _Float16* qh  = xh  + NX;
  _Float16* kh  = qh  + NX;
  _Float16* vt  = kh  + NX;
  _Float16* oh  = vt  + NX;
  _Float16* wqh = oh  + NX;
  _Float16* wkh = wqh + NW;
  _Float16* wvh = wkh + NW;
  _Float16* woh = wvh + NW;
  float*    gout = (float*)(woh + NW);           // 4*16*1024 f32

  convert_kernel<<<2048, 256, 0, stream>>>(x, Wq, Wk, Wv, Wo, xh, wqh, wkh, wvh, woh);
  qkv_gemm<<<2048, 256, 0, stream>>>(xh, wqh, wkh, wvh, bq, bk, bv, qh, kh, vt);
  global_attn<<<4, 256, 0, stream>>>(qh, kh, vt, gout);
  local_attn<<<128, 256, 0, stream>>>(qh, kh, vt, gout, oh);
  out_gemm<<<1024, 256, 0, stream>>>(oh, woh, bo, (float*)d_out);
}
